// GCN_33114197852819
// MI455X (gfx1250) — matrix operand order repaired and checked by reading, not compiled, not run on hardware
//
#include <hip/hip_runtime.h>

typedef __attribute__((ext_vector_type(16))) _Float16 v16h;
typedef __attribute__((ext_vector_type(8)))  _Float16 v8h;
typedef __attribute__((ext_vector_type(4)))  _Float16 v4h;
typedef __attribute__((ext_vector_type(8)))  float    v8f;

#define CAT16(lo, hi) __builtin_shufflevector((lo), (hi), 0,1,2,3,4,5,6,7,8,9,10,11,12,13,14,15)
#define CAT8(lo, hi)  __builtin_shufflevector((lo), (hi), 0,1,2,3,4,5,6,7)

// ---------------- workspace layout (bytes) ----------------
static constexpr size_t AMIX_OFF = 0;                      // 4*512 f16 = 4096 B
static constexpr size_t B1OFF    = 4096;                   // 512 f32
static constexpr size_t B2OFF    = B1OFF + 512 * 4;        // 256 f32
static constexpr size_t B3OFF    = B2OFF + 256 * 4;        // 128 f32
static constexpr size_t B4OFF    = B3OFF + 128 * 4;        // 48 f32 (padded from 40)
static constexpr size_t W1OFF    = 8192;                                   // KC=8,  NT=32
static constexpr size_t W2OFF    = W1OFF + (size_t)8  * 32 * 512 * 2;      // KC=16, NT=16
static constexpr size_t W3OFF    = W2OFF + (size_t)16 * 16 * 512 * 2;      // KC=8,  NT=8
static constexpr size_t W4OFF    = W3OFF + (size_t)8  * 8  * 512 * 2;      // KC=4,  NT=3

// LDS layout
static constexpr int H_ELEMS  = 48 * 536;        // rows=48, max stride = 512+24
static constexpr int S_STRIDE = 72;              // cols 0..47 = support rows, 48..71 zero pad
static constexpr int S_ELEMS  = 512 * S_STRIDE;
static constexpr int SMEM_BYTES = H_ELEMS * 2 + S_ELEMS * 2;  // 51456 + 73728 = 125184

// ---------------- prep 1: mix matrices + padded biases ----------------
__global__ void gcn_prep_small(const float* __restrict__ adj,
                               const float* __restrict__ nv1,
                               const float* __restrict__ nv2,
                               const float* __restrict__ att1, const float* __restrict__ b1,
                               const float* __restrict__ att2, const float* __restrict__ b2,
                               const float* __restrict__ att3, const float* __restrict__ b3,
                               const float* __restrict__ att4, const float* __restrict__ b4,
                               unsigned char* __restrict__ ws)
{
    __shared__ float aS[12][12];
    const int t = threadIdx.x;
    if (t < 12) {
        float row[12];
        float mx = -1e30f;
        for (int j = 0; j < 12; ++j) {
            float s = 0.f;
            for (int k = 0; k < 10; ++k) s += nv1[t * 10 + k] * nv2[k * 12 + j];
            s = fmaxf(s, 0.f);
            row[j] = s;
            mx = fmaxf(mx, s);
        }
        float sum = 0.f;
        for (int j = 0; j < 12; ++j) { row[j] = __expf(row[j] - mx); sum += row[j]; }
        const float inv = 1.0f / sum;
        for (int j = 0; j < 12; ++j) aS[t][j] = adj[t * 12 + j] + row[j] * inv;
    }
    __syncthreads();

    // B-fragment swizzle of Amix^T, zero-padded:
    // [l][lane][j]: lane<16 -> (K=j, N=lane): val = Amix[lane][j] if lane<12 && j<12
    // lane>=16 (K=16..31) -> 0
    const float* atts[4] = {att1, att2, att3, att4};
    _Float16* am = (_Float16*)(ws + AMIX_OFF);
    for (int e = t; e < 4 * 512; e += blockDim.x) {
        const int l = e >> 9, idx = e & 511, lane = idx >> 4, j = idx & 15;
        float v = 0.f;
        if (lane < 12 && j < 12) v = atts[l][lane * 12 + j] * aS[lane][j];
        am[e] = (_Float16)v;
    }
    float* bw = (float*)(ws + B1OFF);
    for (int f = t; f < 512; f += blockDim.x) bw[f] = b1[f];
    bw = (float*)(ws + B2OFF);
    for (int f = t; f < 256; f += blockDim.x) bw[f] = b2[f];
    bw = (float*)(ws + B3OFF);
    for (int f = t; f < 128; f += blockDim.x) bw[f] = b3[f];
    bw = (float*)(ws + B4OFF);
    for (int f = t; f < 48;  f += blockDim.x) bw[f] = (f < 40) ? b4[f] : 0.f;
}

// ---------------- prep 2: weight swizzle into WMMA B-fragment layout ----------------
// per (kc,nt) tile, per lane 16 f16: j<8 -> W[k0+j][n], j>=8 -> W[k0+16+(j-8)][n],
// k0 = kc*32 + (lane>=16 ? 8 : 0), n = nt*16 + (lane&15)
__global__ void gcn_prep_w(const float* __restrict__ W1, const float* __restrict__ W2,
                           const float* __restrict__ W3, const float* __restrict__ W4,
                           unsigned char* __restrict__ ws)
{
    const float* Wsrc[4] = {W1, W2, W3, W4};
    const int    Kd[4]  = {256, 512, 256, 128};
    const int    Nr[4]  = {512, 256, 128, 40};
    const int    NTa[4] = {32, 16, 8, 3};
    const size_t off[4] = {W1OFF, W2OFF, W3OFF, W4OFF};

    for (int l = 0; l < 4; ++l) {
        const int KC = Kd[l] >> 5, NT = NTa[l], Nreal = Nr[l];
        const int total = KC * NT * 512;
        _Float16* dst = (_Float16*)(ws + off[l]);
        for (int e = blockIdx.x * blockDim.x + threadIdx.x; e < total;
             e += gridDim.x * blockDim.x) {
            const int j = e & 15, lane = (e >> 4) & 31, tile = e >> 9;
            const int nt = tile % NT, kc = tile / NT;
            const int k = kc * 32 + ((j >> 3) << 4) + ((lane >> 4) << 3) + (j & 7);
            const int n = nt * 16 + (lane & 15);
            float v = (n < Nreal) ? Wsrc[l][(size_t)k * Nreal + n] : 0.f;
            dst[e] = (_Float16)v;
        }
    }
}

// ---------------- fused GCN: 4 layers, 4 samples (48 rows) per block ----------------
__global__ __launch_bounds__(512) void gcn_fused(const float* __restrict__ x,
                                                 float* __restrict__ out,
                                                 const unsigned char* __restrict__ ws)
{
    extern __shared__ __align__(16) char smem[];
    _Float16* __restrict__ H = (_Float16*)smem;                  // 48 x (Fin+24), row-major
    _Float16* __restrict__ S = (_Float16*)(smem + H_ELEMS * 2);  // Fout x 72, feat-major

    const int tid  = threadIdx.x;
    const int lane = tid & 31;
    const int wave = tid >> 5;
    const int hl   = lane >> 4;
    const int l15  = lane & 15;
    const int gs0  = blockIdx.x * 4;

    // zero S once (cols 48..71 must stay finite-zero forever; K>=12 pad hits them)
    for (int t = tid; t < (S_ELEMS >> 1); t += 512) ((unsigned int*)S)[t] = 0u;

    // load X tile: 4 samples x 12 x 256 f32 -> H f16 (row stride 280)
    {
        const float4* xg = (const float4*)(x + (size_t)gs0 * 12 * 256);
        for (int t = tid; t < 3072; t += 512) {
            float4 v = xg[t];
            const int e = t * 4;
            const int r = e >> 8, c = e & 255;
            v4h h = {(_Float16)v.x, (_Float16)v.y, (_Float16)v.z, (_Float16)v.w};
            *(v4h*)(&H[r * 280 + c]) = h;
        }
    }

    const int    FINa[4]  = {256, 512, 256, 128};
    const int    FOUTa[4] = {512, 256, 128, 48};
    const size_t WOFFa[4] = {W1OFF, W2OFF, W3OFF, W4OFF};
    const size_t BOFFa[4] = {B1OFF, B2OFF, B3OFF, B4OFF};

#pragma unroll
    for (int l = 0; l < 4; ++l) {
        const int Fin = FINa[l], Fout = FOUTa[l];
        const int KC = Fin >> 5, NT = Fout >> 4;
        const int Hs = Fin + 24;
        const _Float16* __restrict__ Wz = (const _Float16*)(ws + WOFFa[l]);
        const float*    __restrict__ bl = (const float*)(ws + BOFFa[l]);

        __syncthreads();
        // ---- GEMM: support = H @ W ; store feat-major, compact cols (col = row r)
        if (l < 3) {
            const int NTH = NT >> 1;          // pairs of N-tiles share one A fragment
            for (int p = wave; p < 3 * NTH; p += 16) {
                const int mt  = p / NTH;
                const int nt0 = (p - mt * NTH) * 2;
                v8f acc0 = {}, acc1 = {};
                const int arow = mt * 16 + l15;
                const int koff = hl * 8;
#pragma unroll
                for (int kc = 0; kc < KC; ++kc) {
                    v8h alo = *(const v8h*)(&H[arow * Hs + kc * 32 + koff]);
                    v8h ahi = *(const v8h*)(&H[arow * Hs + kc * 32 + koff + 16]);
                    v16h A = CAT16(alo, ahi);
                    const v8h* wp = (const v8h*)(Wz + ((size_t)(kc * NT + nt0) * 32 + lane) * 16);
                    v16h B0 = CAT16(wp[0], wp[1]);
                    v16h B1 = CAT16(wp[64], wp[65]);   // next N-tile: +512 f16
                    acc0 = __builtin_amdgcn_wmma_f32_16x16x32_f16(false, A, false, B0,
                                                                  (short)0, acc0, false, false);
                    acc1 = __builtin_amdgcn_wmma_f32_16x16x32_f16(false, A, false, B1,
                                                                  (short)0, acc1, false, false);
                }
                // D: lane = col f, 8 consecutive support-rows rb..rb+7 -> one b128 store
                const int f0 = nt0 * 16 + l15;
                const int rb = mt * 16 + hl * 8;
                v8h h0, h1;
#pragma unroll
                for (int i = 0; i < 8; ++i) h0[i] = (_Float16)acc0[i];
#pragma unroll
                for (int i = 0; i < 8; ++i) h1[i] = (_Float16)acc1[i];
                *(v8h*)(&S[f0 * S_STRIDE + rb])        = h0;
                *(v8h*)(&S[(f0 + 16) * S_STRIDE + rb]) = h1;
            }
        } else {
            for (int t = wave; t < 3 * NT; t += 16) {
                const int mt = t / NT, nt = t - mt * NT;
                v8f acc = {};
                const int arow = mt * 16 + l15;
                const int koff = hl * 8;
#pragma unroll
                for (int kc = 0; kc < KC; ++kc) {
                    v8h alo = *(const v8h*)(&H[arow * Hs + kc * 32 + koff]);
                    v8h ahi = *(const v8h*)(&H[arow * Hs + kc * 32 + koff + 16]);
                    v16h A = CAT16(alo, ahi);
                    const v8h* wp = (const v8h*)(Wz + ((size_t)(kc * NT + nt) * 32 + lane) * 16);
                    v16h B = CAT16(wp[0], wp[1]);
                    acc = __builtin_amdgcn_wmma_f32_16x16x32_f16(false, A, false, B,
                                                                 (short)0, acc, false, false);
                }
                const int f  = nt * 16 + l15;
                const int rb = mt * 16 + hl * 8;
                v8h hv;
#pragma unroll
                for (int i = 0; i < 8; ++i) hv[i] = (_Float16)acc[i];
                *(v8h*)(&S[f * S_STRIDE + rb]) = hv;
            }
        }
        __syncthreads();
        // ---- MIX: out^T[f,n] = sum_k S^T[f,k] * Amix^T[k,n] + b[f]; relu
        {
            const v8h* mp = (const v8h*)((const _Float16*)(ws + AMIX_OFF) +
                                         (size_t)l * 512 + (size_t)lane * 16);
            v16h Bm = CAT16(mp[0], mp[1]);
            for (int u = wave; u < 4 * NT; u += 16) {
                const int s = u & 3, fc = u >> 2;
                const int fr = fc * 16 + l15;          // A row = feature
                const int cb = s * 12 + hl * 16;       // A K run base (8B aligned)
                const _Float16* sp = &S[fr * S_STRIDE + cb];
                v4h a0 = *(const v4h*)(sp);
                v4h a1 = *(const v4h*)(sp + 4);
                v4h a2 = *(const v4h*)(sp + 8);
                v4h a3 = *(const v4h*)(sp + 12);
                v16h A = CAT16(CAT8(a0, a1), CAT8(a2, a3));
                const int fb = fc * 16 + hl * 8;       // D rows = features fb..fb+7
                const float4* bp = (const float4*)(bl + fb);
                float4 c0 = bp[0], c1 = bp[1];
                v8f c = {c0.x, c0.y, c0.z, c0.w, c1.x, c1.y, c1.z, c1.w};
                v8f d = __builtin_amdgcn_wmma_f32_16x16x32_f16(false, A, false, Bm,
                                                               (short)0, c, false, false);
                const int nd = l15;                    // D col = node
                if (l < 3) {
                    if (nd < 12) {
                        v8h hv;
#pragma unroll
                        for (int i = 0; i < 8; ++i) hv[i] = (_Float16)fmaxf(d[i], 0.0f);
                        *(v8h*)(&H[(s * 12 + nd) * (Fout + 24) + fb]) = hv;
                    }
                } else {
                    if (nd < 12 && fb < 40) {          // fb==40 slice is all padding
                        float* op = out + ((size_t)(gs0 + s) * 12 + nd) * 40 + fb;
                        *(float4*)(op)     = make_float4(fmaxf(d[0], 0.f), fmaxf(d[1], 0.f),
                                                         fmaxf(d[2], 0.f), fmaxf(d[3], 0.f));
                        *(float4*)(op + 4) = make_float4(fmaxf(d[4], 0.f), fmaxf(d[5], 0.f),
                                                         fmaxf(d[6], 0.f), fmaxf(d[7], 0.f));
                    }
                }
            }
        }
    }
}

extern "C" void kernel_launch(void* const* d_in, const int* in_sizes, int n_in,
                              void* d_out, int out_size, void* d_ws, size_t ws_size,
                              hipStream_t stream)
{
    (void)n_in; (void)out_size; (void)ws_size;
    const float* x    = (const float*)d_in[0];
    const float* adj  = (const float*)d_in[1];
    const float* nv1  = (const float*)d_in[2];
    const float* nv2  = (const float*)d_in[3];
    const float* W1   = (const float*)d_in[4];
    const float* att1 = (const float*)d_in[5];
    const float* b1   = (const float*)d_in[6];
    const float* W2   = (const float*)d_in[7];
    const float* att2 = (const float*)d_in[8];
    const float* b2   = (const float*)d_in[9];
    const float* W3   = (const float*)d_in[10];
    const float* att3 = (const float*)d_in[11];
    const float* b3   = (const float*)d_in[12];
    const float* W4   = (const float*)d_in[13];
    const float* att4 = (const float*)d_in[14];
    const float* b4   = (const float*)d_in[15];
    unsigned char* ws = (unsigned char*)d_ws;
    float* out = (float*)d_out;

    hipFuncSetAttribute((const void*)gcn_fused,
                        hipFuncAttributeMaxDynamicSharedMemorySize, SMEM_BYTES);

    const int Bn = in_sizes[0] / (12 * 256);   // 65536
    gcn_prep_small<<<1, 256, 0, stream>>>(adj, nv1, nv2, att1, b1, att2, b2,
                                          att3, b3, att4, b4, ws);
    gcn_prep_w<<<256, 256, 0, stream>>>(W1, W2, W3, W4, ws);
    gcn_fused<<<Bn / 4, 512, SMEM_BYTES, stream>>>(x, out, ws);
}
